// ConvNextBlock_16518444220535
// MI455X (gfx1250) — matrix operand order repaired and checked by reading, not compiled, not run on hardware
//
#include <hip/hip_runtime.h>
#include <hip/hip_bf16.h>
#include <math.h>

typedef __attribute__((ext_vector_type(16))) _Float16 v16h;
typedef __attribute__((ext_vector_type(8)))  float    v8f;

#define N_VOX   100000
#define C_CH    64
#define K_TAPS  343
#define HID     256
#define M_TILE  32
#define LDA     72     // padded f16 row stride for hA
#define LDG     264    // padded f16 row stride for hG

// Fragment-swizzled weight sizes (halves): tiles of 32x16, 512 halves per tile
#define W2_TILES (2 * 16)   // ktiles x ntiles for 64x256
#define W3_TILES (8 * 4)    // ktiles x ntiles for 256x64
#define W2H_ELTS (W2_TILES * 512)
#define W3H_ELTS (W3_TILES * 512)

// ---------------------------------------------------------------------------
// Prep: swizzle f32 weight matrix W[k][n] (row stride ldn) into per-lane WMMA
// B-fragment layout, f16. Tile (kt,nt) covers K=kt*32..+31, N=nt*16..+15.
// Within a tile: element t = lane*16 + e, lane: n=lane&15, hi=lane>>4,
// e in 0..15 -> K = kt*32 + (e < 8 ? hi*8 + e : 16 + hi*8 + (e-8)) (matches A fragment order).
// ---------------------------------------------------------------------------
__global__ void swizzle_weights(const float* __restrict__ W,
                                _Float16* __restrict__ Wsw,
                                int ldn, int ntiles_n, int total) {
  const int t = blockIdx.x * blockDim.x + threadIdx.x;
  if (t >= total) return;
  const int tile = t >> 9;
  const int rpos = t & 511;
  const int lane = rpos >> 4;
  const int e    = rpos & 15;
  const int kt   = tile / ntiles_n;
  const int nt   = tile % ntiles_n;
  const int n    = lane & 15;
  const int hi   = lane >> 4;
  const int k    = kt * 32 + ((e < 8) ? (hi * 8 + e) : (16 + hi * 8 + (e - 8)));
  Wsw[t] = (_Float16)W[(long)k * ldn + nt * 16 + n];
}

// A-matrix 16x32 f16 from LDS (row-major). Two contiguous 8-half chunks per
// lane -> compiler emits 2x ds_load_b128 (verified in round-1 disasm).
__device__ __forceinline__ v16h load_a_lds(const _Float16* base, int ld,
                                           int kbase, int lane) {
  const int m  = lane & 15;
  const int hi = (lane >> 4) & 1;
  v16h a;
#pragma unroll
  for (int j = 0; j < 8; ++j) {
    const int k = (j < 4) ? (hi * 8 + 2 * j) : (16 + hi * 8 + 2 * (j - 4));
    a[2 * j]     = base[m * ld + kbase + k];
    a[2 * j + 1] = base[m * ld + kbase + k + 1];
  }
  return a;
}

// B-fragment from pre-swizzled f16 weights: one 32-byte contiguous read.
__device__ __forceinline__ v16h load_b_sw(const _Float16* __restrict__ Wsw,
                                          int tile, int lane) {
  return *(const v16h*)(Wsw + (long)tile * 512 + lane * 16);
}

__global__ __launch_bounds__(256)
void convnext_sparse_block(const float* __restrict__ x_feat,
                           const int*   __restrict__ nbr,
                           const float* __restrict__ W1,
                           const float* __restrict__ b1,
                           const float* __restrict__ gamma,
                           const float* __restrict__ beta,
                           const _Float16* __restrict__ W2h,
                           const float* __restrict__ b2,
                           const _Float16* __restrict__ W3h,
                           const float* __restrict__ b3,
                           float* __restrict__ out) {
  __shared__ int      snbr[M_TILE][64];     // staged neighbor indices (8 KB)
  __shared__ _Float16 hA[M_TILE][LDA];      // GroupNorm output, f16 (4.6 KB)
  __shared__ _Float16 hG[M_TILE][LDG];      // GELU output, f16 (16.9 KB)
  __shared__ float    psum[M_TILE][8];
  __shared__ float    psq [M_TILE][8];

  const int tid  = threadIdx.x;
  const int lane = tid & 31;
  const int wave = tid >> 5;
  const int base = blockIdx.x * M_TILE;

  // ---------------- Phase A: depthwise 7^3 submanifold gather conv ---------
  const int r   = tid >> 3;          // voxel row within tile (0..31)
  const int c0  = (tid & 7) * 8;     // this thread's 8-channel slice

  float acc[8];
#pragma unroll
  for (int i = 0; i < 8; ++i) acc[i] = 0.0f;

  for (int kc = 0; kc < K_TAPS; kc += 64) {
    __syncthreads();
    // cooperative, coalesced staging of the dominant nbr stream
    for (int i = tid; i < M_TILE * 64; i += 256) {
      const int rr = i >> 6, kk = i & 63;
      snbr[rr][kk] = (kc + kk < K_TAPS)
                         ? nbr[(long)(base + rr) * K_TAPS + kc + kk]
                         : N_VOX;  // sentinel -> skipped
    }
    __syncthreads();
    // prefetch next chunk of the 137 MB nbr stream while consuming this one
    if (kc + 64 < K_TAPS) {
      __builtin_prefetch(&nbr[(long)(base + r) * K_TAPS + kc + 64 + (tid & 7) * 8],
                         0, 3);
    }
#pragma unroll 4
    for (int k = 0; k < 64; ++k) {
      const int idx = snbr[r][k];
      if (idx < N_VOX) {  // sparse skip (~75% of taps)
        const float4 f0 = *(const float4*)(x_feat + (long)idx * C_CH + c0);
        const float4 f1 = *(const float4*)(x_feat + (long)idx * C_CH + c0 + 4);
        const float4 w0 = *(const float4*)(W1 + (long)(kc + k) * C_CH + c0);
        const float4 w1 = *(const float4*)(W1 + (long)(kc + k) * C_CH + c0 + 4);
        acc[0] += f0.x * w0.x; acc[1] += f0.y * w0.y;
        acc[2] += f0.z * w0.z; acc[3] += f0.w * w0.w;
        acc[4] += f1.x * w1.x; acc[5] += f1.y * w1.y;
        acc[6] += f1.z * w1.z; acc[7] += f1.w * w1.w;
      }
    }
  }

  // ---------------- Phase B: GroupNorm (1 group over 64 channels) ----------
  float s = 0.0f, sq = 0.0f;
#pragma unroll
  for (int i = 0; i < 8; ++i) {
    acc[i] += b1[c0 + i];
    s  += acc[i];
    sq += acc[i] * acc[i];
  }
  psum[r][tid & 7] = s;
  psq [r][tid & 7] = sq;
  __syncthreads();
  float mu = 0.0f, m2 = 0.0f;
#pragma unroll
  for (int i = 0; i < 8; ++i) { mu += psum[r][i]; m2 += psq[r][i]; }
  mu *= (1.0f / 64.0f);
  m2  = m2 * (1.0f / 64.0f) - mu * mu;
  const float rstd = rsqrtf(m2 + 1e-5f);
#pragma unroll
  for (int i = 0; i < 8; ++i) {
    const float y = (acc[i] - mu) * rstd * gamma[c0 + i] + beta[c0 + i];
    hA[r][c0 + i] = (_Float16)y;
  }
  __syncthreads();

  // ---------------- Phase C: GEMM1 (32x64 @ 64x256) + GELU, WMMA f16 ------
  const int mt   = wave >> 2;                 // row tile (0..1)
  const int mrow = (lane >> 4) * 8;           // C/D row base within tile
  const int ncol = lane & 15;                 // C/D column within tile
  const v16h a0 = load_a_lds(&hA[mt * 16][0], LDA, 0,  lane);
  const v16h a1 = load_a_lds(&hA[mt * 16][0], LDA, 32, lane);
#pragma unroll
  for (int t = 0; t < 4; ++t) {
    const int ct = (wave & 3) * 4 + t;        // output col tile (0..15)
    const int nb = ct * 16;
    v8f cacc = {};
    cacc = __builtin_amdgcn_wmma_f32_16x16x32_f16(
        false, a0, false, load_b_sw(W2h, 0 * 16 + ct, lane),
        (short)0, cacc, false, false);
    cacc = __builtin_amdgcn_wmma_f32_16x16x32_f16(
        false, a1, false, load_b_sw(W2h, 1 * 16 + ct, lane),
        (short)0, cacc, false, false);
#pragma unroll
    for (int j = 0; j < 8; ++j) {
      const int mm = mt * 16 + mrow + j;
      const int nn = nb + ncol;
      const float x = cacc[j] + b2[nn];
      const float g = 0.5f * x * (1.0f + erff(x * 0.70710678118654752f));
      hG[mm][nn] = (_Float16)g;
    }
  }
  __syncthreads();

  // ---------------- Phase D: GEMM2 (32x256 @ 256x64) + residual -----------
  const int mt2 = wave >> 2;        // row tile (0..1)
  const int ct2 = wave & 3;         // col tile (0..3)
  v8f dacc = {};
#pragma unroll
  for (int ks = 0; ks < 8; ++ks) {
    const v16h ag = load_a_lds(&hG[mt2 * 16][0], LDG, ks * 32, lane);
    const v16h bg = load_b_sw(W3h, ks * 4 + ct2, lane);
    dacc = __builtin_amdgcn_wmma_f32_16x16x32_f16(
        false, ag, false, bg, (short)0, dacc, false, false);
  }
#pragma unroll
  for (int j = 0; j < 8; ++j) {
    const int  mm   = mt2 * 16 + mrow + j;
    const int  nn   = ct2 * 16 + ncol;
    const long grow = (long)(base + mm);
    out[grow * C_CH + nn] = dacc[j] + b3[nn] + x_feat[grow * C_CH + nn];
  }
}

extern "C" void kernel_launch(void* const* d_in, const int* in_sizes, int n_in,
                              void* d_out, int out_size, void* d_ws, size_t ws_size,
                              hipStream_t stream) {
  (void)in_sizes; (void)n_in; (void)out_size; (void)ws_size;
  const float* x_feat = (const float*)d_in[0];
  const int*   nbr    = (const int*)  d_in[1];
  const float* W1     = (const float*)d_in[2];
  const float* b1     = (const float*)d_in[3];
  const float* gamma  = (const float*)d_in[4];
  const float* beta   = (const float*)d_in[5];
  const float* W2     = (const float*)d_in[6];
  const float* b2     = (const float*)d_in[7];
  const float* W3     = (const float*)d_in[8];
  const float* b3     = (const float*)d_in[9];
  float* out = (float*)d_out;

  _Float16* W2h = (_Float16*)d_ws;             // 16384 halves
  _Float16* W3h = W2h + W2H_ELTS;              // 16384 halves (64 KB total)

  // Prep: fragment-swizzle weights to f16 (runs every call; deterministic).
  swizzle_weights<<<(W2H_ELTS + 255) / 256, 256, 0, stream>>>(W2, W2h, HID, 16,
                                                              W2H_ELTS);
  swizzle_weights<<<(W3H_ELTS + 255) / 256, 256, 0, stream>>>(W3, W3h, C_CH, 4,
                                                              W3H_ELTS);

  dim3 grid(N_VOX / M_TILE);  // 100000 / 32 = 3125, exact -> EXEC all-ones
  dim3 block(256);            // 8 wave32 waves
  convnext_sparse_block<<<grid, block, 0, stream>>>(
      x_feat, nbr, W1, b1, gamma, beta, W2h, b2, W3h, b3, out);
}
